// MambaEncoder_53085795779269
// MI455X (gfx1250) — matrix operand order repaired and checked by reading, not compiled, not run on hardware
//
#include <hip/hip_runtime.h>
#include <hip/hip_bf16.h>
#include <math.h>

// ---------------- problem constants (from the reference) ----------------
#define Bq     2
#define Lq     1024
#define Dq     768
#define DIq    1536
#define Nq     16
#define Kq     4
#define Rq     48
#define FFq    3072
#define DEPTHq 6
#define Mq     (Bq * Lq)          // 2048 rows in every GEMM
#define XPN    (Rq + 2 * Nq)      // 80 columns of the x-projection

typedef __bf16 v16bf_t __attribute__((ext_vector_type(16)));
typedef __bf16 v8bf_t  __attribute__((ext_vector_type(8)));
typedef float  v8f_t   __attribute__((ext_vector_type(8)));

static __device__ __forceinline__ __bf16 f2bf(float f) {
  unsigned int u = __builtin_bit_cast(unsigned int, f);
  unsigned int r = u + 0x7FFFu + ((u >> 16) & 1u);   // round-to-nearest-even
  unsigned short h = (unsigned short)(r >> 16);
  return __builtin_bit_cast(__bf16, h);
}

static __device__ __forceinline__ float silu_f(float x) {
  return x / (1.f + __expf(-x));
}

// ---------------- f32 -> bf16 bulk conversion (weight staging) -------------
__global__ __launch_bounds__(256) void cvt_bf16_kernel(
    const float* __restrict__ src, __bf16* __restrict__ dst, int n4)
{
  const int i = blockIdx.x * blockDim.x + threadIdx.x;   // one float4 each
  if (i >= n4) return;
  const float4 v = reinterpret_cast<const float4*>(src)[i];
  __bf16* p = dst + (size_t)i * 4;
  p[0] = f2bf(v.x); p[1] = f2bf(v.y); p[2] = f2bf(v.z); p[3] = f2bf(v.w);
}

// ---------------------------------------------------------------------------
// WMMA GEMM:  C[M x N] = act( A[M x K] * W[N x K]^T + bias )
//   A, W bf16 row-major; f32 accumulate via v_wmma_f32_16x16x32_bf16.
//   Each wave owns a 16(M) x (16*NT)(N) strip: one A fragment feeds NT WMMAs.
//   NT must divide N/16 exactly (no guards -> EXEC stays all-1s, no exec
//   juggling around WMMA). K loop is software-pipelined with ping-pong
//   fragment buffers so next-step loads overlap current-step WMMAs.
//   rev_len != 0 : read A rows reversed within each segment of rev_len rows.
//   act: 0=none 1=gelu(exact) 2=softplus 3=silu
//   Outputs (all optional): Cf (f32), Cf2 (f32 duplicate), Cbf (bf16 shadow)
// ---------------------------------------------------------------------------
template<int NT>
__global__ __launch_bounds__(256) void wmma_gemm_kernel(
    const __bf16* __restrict__ A, int lda,
    const __bf16* __restrict__ W, int ldw,
    const float* __restrict__ bias,
    float* __restrict__ Cf, float* __restrict__ Cf2, __bf16* __restrict__ Cbf,
    int M, int N, int Kd, int act, int rev_len)
{
  const int lane   = threadIdx.x & 31;
  const int wave   = threadIdx.x >> 5;
  const int tilesN = N >> 4;
  const int tilesM = M >> 4;
  const int nstrip = tilesN / NT;            // exact by construction
  const int strip  = blockIdx.x * 8 + wave;
  if (strip >= tilesM * nstrip) return;
  const int tm  = strip / nstrip;
  const int tn0 = (strip % nstrip) * NT;
  const int lhalf = lane >> 4;   // 0: lanes 0-15, 1: lanes 16-31
  const int lrow  = lane & 15;

  // A fragment row for this lane (M = lane&15 for both half-waves)
  int am = tm * 16 + lrow;
  if (rev_len) {                 // backward direction: read flipped sequence
    const int b = am / rev_len, l = am % rev_len;
    am = b * rev_len + (rev_len - 1 - l);
  }
  const __bf16* __restrict__ Arow = A + (size_t)am * lda;
  const __bf16* Wr[NT];
#pragma unroll
  for (int j = 0; j < NT; ++j)
    Wr[j] = W + (size_t)((tn0 + j) * 16 + lrow) * ldw;

  // A 16x32 fragment: per-lane two contiguous 8-elem (16B) chunks
  //   elems 0..7  -> K = k      + lhalf*8 + e
  //   elems 8..15 -> K = k + 16 + lhalf*8 + (e-8)
  auto loadA = [&](int kk) -> v16bf_t {
    const v8bf_t a0 = *reinterpret_cast<const v8bf_t*>(Arow + kk + lhalf * 8);
    const v8bf_t a1 = *reinterpret_cast<const v8bf_t*>(Arow + kk + 16 + lhalf * 8);
    return __builtin_shufflevector(a0, a1,
        0,1,2,3,4,5,6,7,8,9,10,11,12,13,14,15);
  };
  // B 32x16 fragment: same K order as A (two 8-elem chunks at k+lhalf*8, k+16+lhalf*8)
  auto loadB = [&](int j, int kk) -> v16bf_t {
    const v8bf_t b0 = *reinterpret_cast<const v8bf_t*>(Wr[j] + kk + lhalf * 8);
    const v8bf_t b1 = *reinterpret_cast<const v8bf_t*>(Wr[j] + kk + 16 + lhalf * 8);
    return __builtin_shufflevector(b0, b1,
        0,1,2,3,4,5,6,7,8,9,10,11,12,13,14,15);
  };

  v8f_t acc[NT];
#pragma unroll
  for (int j = 0; j < NT; ++j) acc[j] = v8f_t{0.f,0.f,0.f,0.f,0.f,0.f,0.f,0.f};

  v16bf_t a0buf, a1buf, b0buf[NT], b1buf[NT];
  int k = 0;
  if (k + 32 <= Kd) {                        // prologue: buffer 0 at k
    a0buf = loadA(k);
#pragma unroll
    for (int j = 0; j < NT; ++j) b0buf[j] = loadB(j, k);
  }
  while (k + 32 <= Kd) {                     // invariant: buf0 holds frags @ k
    const int k2 = k + 32;
    const bool more = (k2 + 32 <= Kd);
    if (more) {                              // issue next loads before WMMAs
      a1buf = loadA(k2);
#pragma unroll
      for (int j = 0; j < NT; ++j) b1buf[j] = loadB(j, k2);
      __builtin_prefetch(Arow + k2 + 32 + lhalf * 8, 0, 1);
      __builtin_prefetch(Wr[0] + k2 + 32 + lhalf * 16, 0, 1);
    }
#pragma unroll
    for (int j = 0; j < NT; ++j)
      acc[j] = __builtin_amdgcn_wmma_f32_16x16x32_bf16(
          false, a0buf, false, b0buf[j], (short)0, acc[j], false, false);
    if (!more) { k = k2; break; }

    const int k3 = k2 + 32;
    const bool more2 = (k3 + 32 <= Kd);
    if (more2) {
      a0buf = loadA(k3);
#pragma unroll
      for (int j = 0; j < NT; ++j) b0buf[j] = loadB(j, k3);
      __builtin_prefetch(Arow + k3 + 32 + lhalf * 8, 0, 1);
      __builtin_prefetch(Wr[0] + k3 + 32 + lhalf * 16, 0, 1);
    }
#pragma unroll
    for (int j = 0; j < NT; ++j)
      acc[j] = __builtin_amdgcn_wmma_f32_16x16x32_bf16(
          false, a1buf, false, b1buf[j], (short)0, acc[j], false, false);
    if (!more2) { k = k3; break; }
    k = k3;
  }
  if (k < Kd) {                              // zero-padded K tail (Kd = 48)
    v16bf_t af;
#pragma unroll
    for (int e = 0; e < 16; ++e) {
      const int ka = k + (e >> 3) * 16 + lhalf * 8 + (e & 7);
      af[e] = (ka < Kd) ? Arow[ka] : (__bf16)0.0f;
    }
#pragma unroll
    for (int j = 0; j < NT; ++j) {
      v16bf_t bf;
#pragma unroll
      for (int e = 0; e < 16; ++e) {
        const int kb = k + (e >> 3) * 16 + lhalf * 8 + (e & 7);
        bf[e] = (kb < Kd) ? Wr[j][kb] : (__bf16)0.0f;
      }
      acc[j] = __builtin_amdgcn_wmma_f32_16x16x32_bf16(
          false, af, false, bf, (short)0, acc[j], false, false);
    }
  }

#pragma unroll
  for (int j = 0; j < NT; ++j) {
    const int   cn = (tn0 + j) * 16 + lrow;
    const float bv = bias ? bias[cn] : 0.f;
#pragma unroll
    for (int r = 0; r < 8; ++r) {  // C/D: VGPR r, half lhalf -> row r + 8*lhalf
      float v = acc[j][r] + bv;
      if (act == 1)      v = 0.5f * v * (1.f + erff(v * 0.70710678118654752f));
      else if (act == 2) v = (v > 20.f) ? v : log1pf(__expf(v));
      else if (act == 3) v = silu_f(v);
      const size_t idx = (size_t)(tm * 16 + 8 * lhalf + r) * N + cn;
      if (Cf)  Cf[idx]  = v;
      if (Cf2) Cf2[idx] = v;
      if (Cbf) Cbf[idx] = f2bf(v);
    }
  }
}

// ---------------- depthwise causal conv (K=4) + bias + SiLU ----------------
// xz: [B*L, 2*DI] f32 (cols 0..DI-1 conv input) -> xc f32 + bf16 shadow
__global__ __launch_bounds__(256) void conv_silu_kernel(
    const float* __restrict__ xz, const float* __restrict__ w,
    const float* __restrict__ bias, float* __restrict__ xc,
    __bf16* __restrict__ xcbf)
{
  const int idx = blockIdx.x * blockDim.x + threadIdx.x;
  if (idx >= Bq * Lq * DIq) return;
  const int c  = idx % DIq;
  const int ml = idx / DIq;       // b*L + l
  const int l  = ml % Lq;
  float acc = bias[c];
#pragma unroll
  for (int j = 0; j < Kq; ++j) {
    const int ls = l - (Kq - 1) + j;
    if (ls >= 0)
      acc += w[c * Kq + j] * xz[(size_t)(ml - (Kq - 1) + j) * (2 * DIq) + c];
  }
  const float v = silu_f(acc);
  xc[idx]   = v;
  xcbf[idx] = f2bf(v);
}

// ---------------- LayerNorm (optionally fused residual add) ----------------
// grid = rows (B*L), 256 threads; D = 768 = 3*256. LN output only feeds
// GEMMs, so it is emitted directly in bf16.
__global__ __launch_bounds__(256) void ln_kernel(
    const float* __restrict__ x, const float* __restrict__ add,
    const float* __restrict__ w, const float* __restrict__ bvec,
    float* __restrict__ res_out, __bf16* __restrict__ ln_out)
{
  __shared__ float s1[256], s2[256];
  const int row = blockIdx.x;
  const int tid = threadIdx.x;
  float vals[3], lsum = 0.f, lsq = 0.f;
#pragma unroll
  for (int i = 0; i < 3; ++i) {
    const int d = tid + i * 256;
    float v = x[(size_t)row * Dq + d];
    if (add) v += add[(size_t)row * Dq + d];
    vals[i] = v; lsum += v; lsq += v * v;
  }
  s1[tid] = lsum; s2[tid] = lsq;
  __syncthreads();
  for (int s = 128; s > 0; s >>= 1) {
    if (tid < s) { s1[tid] += s1[tid + s]; s2[tid] += s2[tid + s]; }
    __syncthreads();
  }
  const float mean = s1[0] * (1.f / Dq);
  const float var  = s2[0] * (1.f / Dq) - mean * mean;
  const float rstd = rsqrtf(var + 1e-5f);
#pragma unroll
  for (int i = 0; i < 3; ++i) {
    const int d = tid + i * 256;
    if (res_out) res_out[(size_t)row * Dq + d] = vals[i];
    ln_out[(size_t)row * Dq + d] = f2bf((vals[i] - mean) * rstd * w[d] + bvec[d]);
  }
}

// ---------------- selective scan: sequential over L, 1 thread / channel ----
// h[N], A[N] live in VGPRs; output only feeds the out-proj GEMM -> bf16.
__global__ __launch_bounds__(256) void scan_kernel(
    const float* __restrict__ dt,    // [B*L, DI]
    const float* __restrict__ xdbl,  // [B*L, 80] : [dtlr | Bs | Cs]
    const float* __restrict__ xc,    // [B*L, DI]
    const float* __restrict__ xz,    // [B*L, 2*DI], z = cols DI..
    const float* __restrict__ A_log, // [DI, N]
    const float* __restrict__ Dp,    // [DI]
    __bf16* __restrict__ ybf)        // [B*L, DI]
{
  const int di = blockIdx.x * blockDim.x + threadIdx.x;
  const int b  = blockIdx.y;
  if (di >= DIq) return;
  float Av[Nq], hst[Nq];
#pragma unroll
  for (int n = 0; n < Nq; ++n) {
    Av[n]  = -__expf(A_log[(size_t)di * Nq + n]);
    hst[n] = 0.f;
  }
  const float dpar = Dp[di];
  for (int l = 0; l < Lq; ++l) {
    const size_t m   = (size_t)b * Lq + l;
    const float dtv  = dt[m * DIq + di];
    const float xv   = xc[m * DIq + di];
    const float zv   = xz[m * (2 * DIq) + DIq + di];
    const float* bc  = xdbl + m * XPN + Rq;   // Bs[0..15], Cs[16..31]
    const float dtx  = dtv * xv;
    float y = 0.f;
#pragma unroll
    for (int n = 0; n < Nq; ++n) {
      const float dA = __expf(dtv * Av[n]);
      hst[n] = dA * hst[n] + dtx * bc[n];
      y += hst[n] * bc[Nq + n];
    }
    ybf[m * DIq + di] = f2bf((y + xv * dpar) * silu_f(zv));
  }
}

// -------- m = 0.5*(out_f + flip(out_r)); residual += m (in place) ----------
__global__ __launch_bounds__(256) void combine_kernel(
    const float* __restrict__ outf, const float* __restrict__ outr,
    float* __restrict__ res)
{
  const int idx = blockIdx.x * blockDim.x + threadIdx.x;
  if (idx >= Bq * Lq * Dq) return;
  const int d  = idx % Dq;
  const int ml = idx / Dq;
  const int l  = ml % Lq;
  const int b  = ml / Lq;
  const size_t ridx = ((size_t)b * Lq + (Lq - 1 - l)) * Dq + d;
  res[idx] += 0.5f * (outf[idx] + outr[ridx]);
}

// ---------------- dt_lr extraction (first R columns of xdbl) ---------------
__global__ __launch_bounds__(256) void extract_dtlr_kernel(
    const float* __restrict__ xdbl, float* __restrict__ dst)
{
  const int idx = blockIdx.x * blockDim.x + threadIdx.x;
  if (idx >= Bq * Lq * Rq) return;
  const int r = idx % Rq;
  const int m = idx / Rq;
  dst[idx] = xdbl[(size_t)m * XPN + r];
}

// ===========================================================================
extern "C" void kernel_launch(void* const* d_in, const int* in_sizes, int n_in,
                              void* d_out, int out_size, void* d_ws, size_t ws_size,
                              hipStream_t stream) {
  (void)in_sizes; (void)n_in; (void)out_size; (void)ws_size;

  const float* x_in     = (const float*)d_in[0];
  const float* ln1_w    = (const float*)d_in[1];
  const float* ln1_b    = (const float*)d_in[2];
  const float* ln2_w    = (const float*)d_in[3];
  const float* ln2_b    = (const float*)d_in[4];
  const float* in_w     = (const float*)d_in[5];
  const float* conv_w   = (const float*)d_in[6];
  const float* conv_b   = (const float*)d_in[7];
  const float* xproj_w  = (const float*)d_in[8];
  const float* dtproj_w = (const float*)d_in[9];
  const float* dtproj_b = (const float*)d_in[10];
  const float* A_log    = (const float*)d_in[11];
  const float* D_param  = (const float*)d_in[12];
  const float* out_w    = (const float*)d_in[13];
  const float* mlp_w1   = (const float*)d_in[14];
  const float* mlp_b1   = (const float*)d_in[15];
  const float* mlp_w2   = (const float*)d_in[16];
  const float* mlp_b2   = (const float*)d_in[17];

  // ---- outputs: [x | xs | dts | dtlr], concatenated flat (all f32) ----
  const size_t BLD = (size_t)Bq * Lq * Dq;
  float* out       = (float*)d_out;
  float* out_x     = out;
  float* out_xs    = out + BLD;
  float* out_dts   = out_xs + (size_t)DEPTHq * BLD;
  float* out_dtlr  = out_dts + (size_t)DEPTHq * 2 * Bq * Lq * DIq;

  // ---- workspace carve-up (~97 MB total; lives in the 192MB L2) ----
  float* ws   = (float*)d_ws;
  float* res  = ws;                                 // [B*L, D]   f32
  float* xz   = res  + BLD;                         // [B*L, 2DI] f32
  float* xc   = xz   + (size_t)Mq * 2 * DIq;        // [B*L, DI]  f32
  float* xdbl = xc   + (size_t)Mq * DIq;            // [B*L, 80]  f32
  float* outf = xdbl + (size_t)Mq * XPN;            // [B*L, D]   f32
  float* outr = outf + BLD;                         // [B*L, D]   f32
  float* xcur = outr + BLD;                         // [B*L, D]   f32
  __bf16* bfbase = (__bf16*)(xcur + BLD);
  __bf16* h_bf    = bfbase;                         // [B*L, D]   bf16
  __bf16* xc_bf   = h_bf   + BLD;                   // [B*L, DI]  bf16
  __bf16* xdbl_bf = xc_bf  + (size_t)Mq * DIq;      // [B*L, 80]  bf16
  __bf16* y_bf    = xdbl_bf + (size_t)Mq * XPN;     // [B*L, DI]  bf16
  __bf16* hid_bf  = y_bf   + (size_t)Mq * DIq;      // [B*L, FF]  bf16
  __bf16* wstage  = hid_bf + (size_t)Mq * FFq;      // weight staging (bf16)

  auto cvtW = [&](const float* src, int n) {
    cvt_bf16_kernel<<<(n / 4 + 255) / 256, 256, 0, stream>>>(src, wstage, n / 4);
  };
  auto gemm = [&](const __bf16* A, int lda, int Kd, const float* bias,
                  float* Cf, float* Cf2, __bf16* Cbf, int N, int act, int rev) {
    const int tilesN = N / 16;
    if (tilesN % 4 == 0) {
      const int waves = (Mq / 16) * (tilesN / 4);
      wmma_gemm_kernel<4><<<(waves + 7) / 8, 256, 0, stream>>>(
          A, lda, wstage, Kd, bias, Cf, Cf2, Cbf, Mq, N, Kd, act, rev);
    } else {               // N = 80 -> tilesN = 5, one 5-tile strip per wave
      const int waves = (Mq / 16) * (tilesN / 5);
      wmma_gemm_kernel<5><<<(waves + 7) / 8, 256, 0, stream>>>(
          A, lda, wstage, Kd, bias, Cf, Cf2, Cbf, Mq, N, Kd, act, rev);
    }
  };
  auto grid1 = [](int count) { return dim3((count + 255) / 256); };

  for (int i = 0; i < DEPTHq; ++i) {
    // residual = x (+ residual); h = LN1(residual) -> bf16
    const float* xsrc = (i == 0) ? x_in : xcur;
    const float* addp = (i == 0) ? nullptr : res;
    ln_kernel<<<Mq, 256, 0, stream>>>(xsrc, addp, ln1_w + i * Dq, ln1_b + i * Dq,
                                      res, h_bf);

    for (int dir = 0; dir < 2; ++dir) {
      const size_t s = (size_t)(i * 2 + dir);
      // in-proj (backward direction reads h with rows flipped per batch)
      cvtW(in_w + s * (2 * DIq) * Dq, 2 * DIq * Dq);
      gemm(h_bf, Dq, Dq, nullptr, xz, nullptr, nullptr, 2 * DIq, 0, dir ? Lq : 0);
      // depthwise causal conv + SiLU
      conv_silu_kernel<<<grid1(Bq * Lq * DIq), 256, 0, stream>>>(
          xz, conv_w + s * DIq * Kq, conv_b + s * DIq, xc, xc_bf);
      // x-proj -> [dt_lr | Bs | Cs] (f32 for scan + bf16 shadow for dt-proj)
      cvtW(xproj_w + s * XPN * DIq, XPN * DIq);
      gemm(xc_bf, DIq, DIq, nullptr, xdbl, nullptr, xdbl_bf, XPN, 0, 0);
      // dt_lr output slice
      extract_dtlr_kernel<<<grid1(Bq * Lq * Rq), 256, 0, stream>>>(
          xdbl, out_dtlr + s * Mq * Rq);
      // dt = softplus(dt_lr @ dtproj_w^T + b) -> straight into d_out
      cvtW(dtproj_w + s * DIq * Rq, DIq * Rq);
      float* dtp = out_dts + s * (size_t)Mq * DIq;
      gemm(xdbl_bf, XPN, Rq, dtproj_b + s * DIq, dtp, nullptr, nullptr,
           DIq, /*softplus*/2, 0);
      // selective scan (sequential over L) + D skip + SiLU(z) gate -> bf16
      scan_kernel<<<dim3(DIq / 256, Bq), 256, 0, stream>>>(
          dtp, xdbl, xc, xz, A_log + s * DIq * Nq, D_param + s * DIq, y_bf);
      // out-proj
      cvtW(out_w + s * Dq * DIq, Dq * DIq);
      gemm(y_bf, DIq, DIq, nullptr, dir ? outr : outf, nullptr, nullptr,
           Dq, 0, 0);
    }

    // residual += 0.5*(out_f + flip(out_r)); h2 = LN2(residual) -> bf16
    combine_kernel<<<grid1(Bq * Lq * Dq), 256, 0, stream>>>(outf, outr, res);
    ln_kernel<<<Mq, 256, 0, stream>>>(res, nullptr, ln2_w + i * Dq, ln2_b + i * Dq,
                                      nullptr, h_bf);
    // MLP: gelu(h2 @ W1^T + b1) @ W2^T + b2  (hidden exists only in bf16)
    cvtW(mlp_w1 + (size_t)i * FFq * Dq, FFq * Dq);
    gemm(h_bf, Dq, Dq, mlp_b1 + i * FFq, nullptr, nullptr, hid_bf,
         FFq, /*gelu*/1, 0);
    cvtW(mlp_w2 + (size_t)i * Dq * FFq, Dq * FFq);
    gemm(hid_bf, FFq, FFq, mlp_b2 + i * Dq, xcur, out_xs + (size_t)i * BLD,
         nullptr, Dq, 0, 0);
  }

  // final x output
  hipMemcpyAsync(out_x, xcur, BLD * sizeof(float),
                 hipMemcpyDeviceToDevice, stream);
}
